// SelectiveSSM_16114717294731
// MI455X (gfx1250) — matrix operand order repaired and checked by reading, not compiled, not run on hardware
//
#include <hip/hip_runtime.h>
#include <cstdint>
#include <cstddef>

// ---------------- problem constants (match reference) ----------------
#define D_MODEL  1024
#define D_STATE  16
#define D_CONV   4
#define D_INNER  2048
#define DT_RANK  64
#define BATCH    4
#define SEQ      2048
#define NTOK     (BATCH * SEQ)          // 8192 tokens

typedef __attribute__((ext_vector_type(16))) __bf16 v16bf;
typedef __attribute__((ext_vector_type(8)))  float  v8f;

union Frag16 { v16bf v; uint4 q[2]; };

// ---------------- f32 -> bf16 conversion (4 elems/thread) ----------------
__global__ __launch_bounds__(256) void cvt_f32_bf16x4_kernel(
    const float4* __restrict__ in, uint2* __restrict__ out, int n4) {
  int i = blockIdx.x * 256 + threadIdx.x;
  if (i < n4) {
    float4 f = in[i];
    union { __bf16 h[4]; uint2 u; } p;
    p.h[0] = (__bf16)f.x; p.h[1] = (__bf16)f.y;
    p.h[2] = (__bf16)f.z; p.h[3] = (__bf16)f.w;
    out[i] = p.u;
  }
}

// ---------------- WMMA NT GEMM ----------------
// C[M,N] = A[M,K] * B[N,K]^T   (both A and B row-major, K contiguous)
// Wave tile: (WMT*16) x (WNT*16). Block: 128 threads = 4 waves in 2x2.
// EPI: 0 = none; 1 = softplus(acc + bias[n])
template<int WMT, int WNT, int EPI>
__global__ __launch_bounds__(128) void wmma_gemm_nt(
    const __bf16* __restrict__ A, int lda,
    const __bf16* __restrict__ B, int ldb,
    float* __restrict__ C, int ldc,
    const float* __restrict__ bias,
    int K)
{
  const int lane = threadIdx.x & 31;
  const int wave = threadIdx.x >> 5;
  const int wm   = wave >> 1;
  const int wn   = wave & 1;
  const int m0   = blockIdx.y * (2 * WMT * 16) + wm * (WMT * 16);
  const int n0   = blockIdx.x * (2 * WNT * 16) + wn * (WNT * 16);
  const int r16  = lane & 15;
  const int hi   = lane >> 4;

  v8f acc[WMT][WNT];
#pragma unroll
  for (int i = 0; i < WMT; ++i)
#pragma unroll
    for (int j = 0; j < WNT; ++j) acc[i][j] = {};

  // Per-lane base pointers for the A/B fragments (this code's contiguous K order, same for A and B).
  // A 16x32 bf16: lane holds row (lane&15); K = hi*16 + {0..15} (contiguous)
  // B 32x16 bf16: lane holds col (lane&15); K = hi*16 + {0..15} (contiguous)
  const __bf16* aRow[WMT];
  const __bf16* bRow[WNT];
#pragma unroll
  for (int i = 0; i < WMT; ++i)
    aRow[i] = A + (size_t)(m0 + i * 16 + r16) * lda + hi * 16;
#pragma unroll
  for (int j = 0; j < WNT; ++j)
    bRow[j] = B + (size_t)(n0 + j * 16 + r16) * ldb + hi * 16;

  for (int k0 = 0; k0 < K; k0 += 32) {
    Frag16 a[WMT], b[WNT];
#pragma unroll
    for (int i = 0; i < WMT; ++i) {
      a[i].q[0] = *(const uint4*)(aRow[i] + k0);        // K = k0+hi*16 .. +7
      a[i].q[1] = *(const uint4*)(aRow[i] + k0 + 8);    // K = k0+hi*16+8 .. +15
      __builtin_prefetch(aRow[i] + k0 + 64, 0, 1);      // speculative, next line
    }
#pragma unroll
    for (int j = 0; j < WNT; ++j) {
      const uint4* p = (const uint4*)(bRow[j] + k0);    // K = k0+hi*16 .. +15
      b[j].q[0] = p[0];
      b[j].q[1] = p[1];
      __builtin_prefetch(bRow[j] + k0 + 64, 0, 1);
    }
#pragma unroll
    for (int i = 0; i < WMT; ++i)
#pragma unroll
      for (int j = 0; j < WNT; ++j)
        acc[i][j] = __builtin_amdgcn_wmma_f32_16x16x32_bf16(
            false, a[i].v, false, b[j].v, (short)0, acc[i][j], false, false);
  }

  // Store: f32 C/D layout — VGPR r -> M = hi*8 + r, N = lane&15
#pragma unroll
  for (int i = 0; i < WMT; ++i) {
#pragma unroll
    for (int j = 0; j < WNT; ++j) {
      const int n = n0 + j * 16 + r16;
      float bv = 0.f;
      if (EPI == 1) bv = bias[n];
#pragma unroll
      for (int r = 0; r < 8; ++r) {
        const int m = m0 + i * 16 + hi * 8 + r;
        float v = acc[i][j][r];
        if (EPI == 1) {
          v += bv;                                   // + dt_proj_b
          v = (v > 20.f) ? v : log1pf(__expf(v));    // softplus
        }
        C[(size_t)m * ldc + n] = v;
      }
    }
  }
}

// ---------------- causal depthwise conv(4) + bias + SiLU ----------------
// xz: (NTOK, 2*D_INNER) f32; reads x half (cols [0,D_INNER)); conv along L
// per (b, d) channel with left pad 3. Writes x_ssm f32 and bf16.
__global__ __launch_bounds__(256) void conv_silu_kernel(
    const float* __restrict__ xz,
    const float* __restrict__ cw, const float* __restrict__ cb,
    float* __restrict__ xs, __bf16* __restrict__ xsbf)
{
  const int idx = blockIdx.x * 256 + threadIdx.x;   // over NTOK*D_INNER
  const int d  = idx & (D_INNER - 1);
  const int bl = idx >> 11;                         // token index b*SEQ + l
  const int l  = bl & (SEQ - 1);

  float s = cb[d];
#pragma unroll
  for (int j = 0; j < D_CONV; ++j) {
    const int ll = l - (D_CONV - 1) + j;
    if (ll >= 0)
      s += cw[d * D_CONV + j] *
           xz[(size_t)(bl - (D_CONV - 1 - j)) * (2 * D_INNER) + d];
  }
  const float v = s * (1.f / (1.f + __expf(-s)));   // SiLU
  xs[idx]   = v;
  xsbf[idx] = (__bf16)v;
}

// ---------------- selective scan (sequential in L) ----------------
// One thread per (b, d) channel; 16 fp32 states in registers.
// Fuses y = scan + D*x, then y *= silu(z); emits bf16 for out_proj GEMM.
__global__ __launch_bounds__(256) void scan_kernel(
    const float* __restrict__ dt,    // (NTOK, D_INNER)
    const float* __restrict__ dbl,   // (NTOK, 96): [0:64)=dt_x, [64:80)=B, [80:96)=C
    const float* __restrict__ xs,    // (NTOK, D_INNER)
    const float* __restrict__ xz,    // (NTOK, 2*D_INNER); z = cols [D_INNER, 2*D_INNER)
    const float* __restrict__ A_log, // (D_INNER, 16)
    const float* __restrict__ Dp,    // (D_INNER)
    __bf16* __restrict__ ybf)        // (NTOK, D_INNER)
{
  const int ch = blockIdx.x * 256 + threadIdx.x;    // 0 .. BATCH*D_INNER-1
  const int d  = ch & (D_INNER - 1);
  const int b  = ch >> 11;

  float Ar[D_STATE];
#pragma unroll
  for (int n = 0; n < D_STATE; ++n) Ar[n] = -__expf(A_log[d * D_STATE + n]);
  const float Dd = Dp[d];

  float h[D_STATE];
#pragma unroll
  for (int n = 0; n < D_STATE; ++n) h[n] = 0.f;

  const float* dtp  = dt  + (size_t)b * SEQ * D_INNER + d;
  const float* xsp  = xs  + (size_t)b * SEQ * D_INNER + d;
  const float* xzp  = xz  + (size_t)b * SEQ * (2 * D_INNER) + D_INNER + d;
  const float* dblp = dbl + (size_t)b * SEQ * 96;
  __bf16*      yp   = ybf + (size_t)b * SEQ * D_INNER + d;

  for (int l = 0; l < SEQ; ++l) {
    const float dtv = dtp[(size_t)l * D_INNER];
    const float xv  = xsp[(size_t)l * D_INNER];
    const float zv  = xzp[(size_t)l * (2 * D_INNER)];
    const float* row = dblp + (size_t)l * 96;
    float y = 0.f;
#pragma unroll
    for (int n = 0; n < D_STATE; ++n) {
      const float hn = __expf(dtv * Ar[n]) * h[n] + (dtv * row[DT_RANK + n]) * xv;
      h[n] = hn;
      y += hn * row[DT_RANK + D_STATE + n];
    }
    const float out = (y + Dd * xv) * (zv * (1.f / (1.f + __expf(-zv))));
    yp[(size_t)l * D_INNER] = (__bf16)out;
  }
}

// ---------------- launch ----------------
extern "C" void kernel_launch(void* const* d_in, const int* in_sizes, int n_in,
                              void* d_out, int out_size, void* d_ws, size_t ws_size,
                              hipStream_t stream) {
  (void)in_sizes; (void)n_in; (void)out_size; (void)ws_size;
  const float* x          = (const float*)d_in[0];
  const float* in_proj_w  = (const float*)d_in[1];
  const float* conv_w     = (const float*)d_in[2];
  const float* conv_b     = (const float*)d_in[3];
  const float* x_proj_w   = (const float*)d_in[4];
  const float* dt_proj_w  = (const float*)d_in[5];
  const float* dt_proj_b  = (const float*)d_in[6];
  const float* A_log      = (const float*)d_in[7];
  const float* D_param    = (const float*)d_in[8];
  const float* out_proj_w = (const float*)d_in[9];
  float* out = (float*)d_out;

  // ---- carve workspace (256B-aligned slabs) ----
  size_t off = 0;
  auto carve = [&](size_t bytes) -> void* {
    off = (off + 255) & ~(size_t)255;
    void* p = (char*)d_ws + off;
    off += bytes;
    return p;
  };
  __bf16* Xbf   = (__bf16*)carve((size_t)NTOK * D_MODEL * 2);          // 16 MB
  __bf16* Wip   = (__bf16*)carve((size_t)2 * D_INNER * D_MODEL * 2);   // 8 MB
  __bf16* Wxp   = (__bf16*)carve((size_t)96 * D_INNER * 2);
  __bf16* Wdt   = (__bf16*)carve((size_t)D_INNER * DT_RANK * 2);
  __bf16* Wop   = (__bf16*)carve((size_t)D_MODEL * D_INNER * 2);       // 4 MB
  float*  XZ    = (float*)carve((size_t)NTOK * 2 * D_INNER * 4);       // 128 MB
  float*  XS    = (float*)carve((size_t)NTOK * D_INNER * 4);           // 64 MB
  __bf16* XSbf  = (__bf16*)carve((size_t)NTOK * D_INNER * 2);          // 32 MB
  float*  DBL   = (float*)carve((size_t)NTOK * 96 * 4);                // 3 MB
  __bf16* DBLbf = (__bf16*)carve((size_t)NTOK * 96 * 2);
  float*  DT    = (float*)carve((size_t)NTOK * D_INNER * 4);           // 64 MB
  __bf16* Ybf   = (__bf16*)carve((size_t)NTOK * D_INNER * 2);          // 32 MB

  auto cvt = [&](const float* src, __bf16* dst, int n) {
    const int n4 = n / 4;
    cvt_f32_bf16x4_kernel<<<(n4 + 255) / 256, 256, 0, stream>>>(
        (const float4*)src, (uint2*)dst, n4);
  };

  // 0) bf16 conversions
  cvt(x,          Xbf, NTOK * D_MODEL);
  cvt(in_proj_w,  Wip, 2 * D_INNER * D_MODEL);
  cvt(x_proj_w,   Wxp, 96 * D_INNER);
  cvt(dt_proj_w,  Wdt, D_INNER * DT_RANK);
  cvt(out_proj_w, Wop, D_MODEL * D_INNER);

  // 1) in_proj: XZ[8192,4096] = X[8192,1024] * Wip[4096,1024]^T
  wmma_gemm_nt<2, 4, 0><<<dim3((2 * D_INNER) / 128, NTOK / 64), 128, 0, stream>>>(
      Xbf, D_MODEL, Wip, D_MODEL, XZ, 2 * D_INNER, nullptr, D_MODEL);

  // 2) causal conv + SiLU -> XS (f32) and XSbf (bf16)
  conv_silu_kernel<<<(NTOK * D_INNER) / 256, 256, 0, stream>>>(
      XZ, conv_w, conv_b, XS, XSbf);

  // 3) x_proj: DBL[8192,96] = XS[8192,2048] * Wxp[96,2048]^T
  wmma_gemm_nt<2, 3, 0><<<dim3(1, NTOK / 64), 128, 0, stream>>>(
      XSbf, D_INNER, Wxp, D_INNER, DBL, 96, nullptr, D_INNER);

  // 4) bf16 copy of dbl (dt_x = first 64 cols, lda = 96)
  cvt(DBL, DBLbf, NTOK * 96);

  // 5) dt_proj + softplus: DT[8192,2048] = softplus(dt_x * Wdt^T + b)
  wmma_gemm_nt<2, 4, 1><<<dim3(D_INNER / 128, NTOK / 64), 128, 0, stream>>>(
      DBLbf, 96, Wdt, DT_RANK, DT, D_INNER, dt_proj_b, DT_RANK);

  // 6) selective scan + D*x + silu(z) gating -> Ybf
  scan_kernel<<<(BATCH * D_INNER) / 256, 256, 0, stream>>>(
      DT, DBL, XS, XZ, A_log, D_param, Ybf);

  // 7) out_proj: out[8192,1024] = Y[8192,2048] * Wop[1024,2048]^T
  wmma_gemm_nt<2, 4, 0><<<dim3(D_MODEL / 128, NTOK / 64), 128, 0, stream>>>(
      Ybf, D_INNER, Wop, D_INNER, out, D_MODEL, nullptr, D_INNER);
}
